// LinearSplineKANNetwork_66718021976513
// MI455X (gfx1250) — matrix operand order repaired and checked by reading, not compiled, not run on hardware
//
#include <hip/hip_runtime.h>
#include <hip/hip_bf16.h>
#include <math.h>

// ---------------------------------------------------------------------------
// Problem constants (from the reference)
// ---------------------------------------------------------------------------
#define BATCH   4096
#define I0      128
#define HID     256
#define OUTW    128
#define NGRID   17
#define GRANGE  2.5f
#define SPACING (2.0f * GRANGE / (NGRID - 1))   // 0.3125
#define LN_EPS  1e-5f
#define K0      (I0  + I0  * NGRID)   // 2304
#define K1      (HID + HID * NGRID)   // 4608

typedef __attribute__((ext_vector_type(16))) __bf16 v16bf;
typedef __attribute__((ext_vector_type(8)))  __bf16 v8bf;
typedef __attribute__((ext_vector_type(8)))  float  v8f;
typedef __attribute__((ext_vector_type(4)))  float  v4f;

__device__ __forceinline__ __bf16 f2bf(float f) {
    union { float f; unsigned u; } c; c.f = f;
    unsigned r = (c.u + 0x7fffu + ((c.u >> 16) & 1u)) >> 16;
    return __builtin_bit_cast(__bf16, (unsigned short)r);
}

__device__ __forceinline__ float silu(float x) {
    return x / (1.0f + __expf(-x));
}

// ---------------------------------------------------------------------------
// 0. zero the double accumulators [se_sum, se_sq, sb_sum, sb_sq]
// ---------------------------------------------------------------------------
__global__ void kan_init(double* acc) {
    if (threadIdx.x < 4) acc[threadIdx.x] = 0.0;
}

// ---------------------------------------------------------------------------
// 1. pack weights -> bf16, N-MAJOR: Wt[o][k], k = [base | i*17+g]
//    (N-major so GEMM LDS tiles + WMMA B-fragments read contiguously)
// ---------------------------------------------------------------------------
__global__ __launch_bounds__(256)
void kan_build_W(const float* __restrict__ base_scale,  // (I,O)
                 const float* __restrict__ spline_w,    // (I,O,17)
                 __bf16* __restrict__ Wt, int I, int O, int Ktot) {
    int idx = blockIdx.x * 256 + threadIdx.x;
    if (idx >= Ktot * O) return;
    int o = idx / Ktot, k = idx % Ktot;
    float v;
    if (k < I) {
        v = base_scale[k * O + o];
    } else {
        int k2 = k - I, i = k2 / NGRID, g = k2 % NGRID;
        v = spline_w[((size_t)i * O + o) * NGRID + g];
    }
    Wt[(size_t)o * Ktot + k] = f2bf(v);   // coalesced over k
}

// ---------------------------------------------------------------------------
// 2. build feature matrix A = [silu(x) | hat-basis] in bf16, row length Ktot
// ---------------------------------------------------------------------------
__global__ __launch_bounds__(256)
void kan_build_A0(const float* __restrict__ x, __bf16* __restrict__ A,
                  int B, int I, int Ktot) {
    int idx = blockIdx.x * 256 + threadIdx.x;     // over B*I
    if (idx >= B * I) return;
    int b = idx / I, i = idx % I;
    float xv = x[idx];
    A[(size_t)b * Ktot + i] = f2bf(silu(xv));
    float u = (xv + GRANGE) / SPACING;
    size_t base = (size_t)b * Ktot + I + (size_t)i * NGRID;
#pragma unroll
    for (int g = 0; g < NGRID; ++g) {
        float bs = fmaxf(1.0f - fabsf(u - (float)g), 0.0f);
        A[base + g] = f2bf(bs);
    }
}

// same, but also emits basis1 (f32) to d_out and accumulates its sum / sumsq
__global__ __launch_bounds__(256)
void kan_build_A1(const float* __restrict__ h, __bf16* __restrict__ A,
                  float* __restrict__ basis_out, double* __restrict__ acc,
                  int B, int I, int Ktot) {
    int idx = blockIdx.x * 256 + threadIdx.x;     // over B*I (exact multiple)
    int t = threadIdx.x;
    double ls = 0.0, lq = 0.0;
    if (idx < B * I) {
        int b = idx / I, i = idx % I;
        float hv = h[idx];
        A[(size_t)b * Ktot + i] = f2bf(silu(hv));
        float u = (hv + GRANGE) / SPACING;
        size_t abase = (size_t)b * Ktot + I + (size_t)i * NGRID;
        size_t obase = (size_t)idx * NGRID;
#pragma unroll
        for (int g = 0; g < NGRID; ++g) {
            float bs = fmaxf(1.0f - fabsf(u - (float)g), 0.0f);
            A[abase + g] = f2bf(bs);
            __builtin_nontemporal_store(bs, &basis_out[obase + g]);
            ls += (double)bs;
            lq += (double)bs * (double)bs;
        }
    }
    __shared__ double rs[256];
    __shared__ double rq[256];
    rs[t] = ls; rq[t] = lq;
    __syncthreads();
    for (int off = 128; off > 0; off >>= 1) {
        if (t < off) { rs[t] += rs[t + off]; rq[t] += rq[t + off]; }
        __syncthreads();
    }
    if (t == 0) {
        atomicAdd(&acc[2], rs[0]);
        atomicAdd(&acc[3], rq[0]);
    }
}

// ---------------------------------------------------------------------------
// 3. bf16 WMMA GEMM: C(MxN,f32) = scale * A(MxK,bf16) @ Wt(NxK,bf16)^T + bias
//    block tile 64x64, 8 waves (4 in M x 2 in N), each wave 16x32 via 2 WMMAs
//    double-buffered LDS, one barrier per K-step, all LDS traffic in b128
// ---------------------------------------------------------------------------
__global__ __launch_bounds__(256)
void kan_gemm_bf16(const __bf16* __restrict__ A, const __bf16* __restrict__ Wt,
                   float* __restrict__ C, const float* __restrict__ bias,
                   int M, int N, int K, float scale) {
    __shared__ __bf16 As[2][64][40];   // [buf][m][k]  row stride 80B
    __shared__ __bf16 Bs[2][64][40];   // [buf][n][k]  row stride 80B

    const int bm = blockIdx.x * 64;
    const int bn = blockIdx.y * 64;
    const int tid  = threadIdx.x;
    const int wave = tid >> 5;
    const int lane = tid & 31;
    const int wm = (wave & 3) * 16;   // wave M offset in tile
    const int wn = (wave >> 2) * 32;  // wave N offset in tile

    // global->LDS load mapping: one 16B chunk per thread per tile
    const int arow = tid >> 2, kcol = (tid & 3) * 8;   // A: 64 m-rows x 32 k
    const int orow = tid >> 2;                          // B: 64 n-rows x 32 k

    v8f acc0 = {}; v8f acc1 = {};

    const int mrow = wm + (lane & 15);
    const int klo  = (lane < 16) ? 0 : 8;    // A frag: K {klo..+7, klo+16..+23}
    const int kb   = (lane < 16) ? 0 : 8;    // B frag: K {kb..+7, kb+16..+23}
    const int nc   = lane & 15;

    const int nt = K >> 5;                    // K divisible by 32
    const size_t arowoff = (size_t)(bm + arow) * K + kcol;
    const size_t browoff = (size_t)(bn + orow) * K + kcol;

    // prologue: tile 0 into registers
    v8bf ar = *(const v8bf*)&A [arowoff];
    v8bf br = *(const v8bf*)&Wt[browoff];

    for (int t = 0; t < nt; ++t) {
        const int buf = t & 1;
        *(v8bf*)&As[buf][arow][kcol] = ar;
        *(v8bf*)&Bs[buf][orow][kcol] = br;
        __syncthreads();

        if (t + 1 < nt) {
            ar = *(const v8bf*)&A [arowoff + (size_t)(t + 1) * 32];
            br = *(const v8bf*)&Wt[browoff + (size_t)(t + 1) * 32];
        }
        if (t + 2 < nt) {   // global_prefetch_b8 for tile t+2
            __builtin_prefetch(&A [arowoff + (size_t)(t + 2) * 32], 0, 3);
            __builtin_prefetch(&Wt[browoff + (size_t)(t + 2) * 32], 0, 3);
        }

        // fragments: all reads are contiguous 16B LDS loads
        v8bf alo = *(const v8bf*)&As[buf][mrow][klo];
        v8bf ahi = *(const v8bf*)&As[buf][mrow][klo + 16];
        v16bf af = __builtin_shufflevector(alo, ahi,
                     0,1,2,3,4,5,6,7,8,9,10,11,12,13,14,15);
        v8bf b0l = *(const v8bf*)&Bs[buf][wn + nc][kb];
        v8bf b0h = *(const v8bf*)&Bs[buf][wn + nc][kb + 16];
        v16bf bf0 = __builtin_shufflevector(b0l, b0h,
                     0,1,2,3,4,5,6,7,8,9,10,11,12,13,14,15);
        v8bf b1l = *(const v8bf*)&Bs[buf][wn + 16 + nc][kb];
        v8bf b1h = *(const v8bf*)&Bs[buf][wn + 16 + nc][kb + 16];
        v16bf bf1 = __builtin_shufflevector(b1l, b1h,
                     0,1,2,3,4,5,6,7,8,9,10,11,12,13,14,15);

        acc0 = __builtin_amdgcn_wmma_f32_16x16x32_bf16(
                   false, af, false, bf0, (short)0, acc0, false, false);
        acc1 = __builtin_amdgcn_wmma_f32_16x16x32_bf16(
                   false, af, false, bf1, (short)0, acc1, false, false);
        // no second barrier: next iteration writes the other buffer, whose
        // readers finished before this iteration's barrier
    }

    // C/D layout: VGPR r -> M = r (lanes 0-15) / r+8 (lanes 16-31), N = lane&15
    const int row0 = bm + wm + ((lane < 16) ? 0 : 8);
    const int col0 = bn + wn + nc;
    const float b0 = bias[col0];
    const float b1 = bias[col0 + 16];
#pragma unroll
    for (int r = 0; r < 8; ++r) {
        C[(size_t)(row0 + r) * N + col0]      = acc0[r] * scale + b0;
        C[(size_t)(row0 + r) * N + col0 + 16] = acc1[r] * scale + b1;
    }
}

// ---------------------------------------------------------------------------
// 4. LayerNorm over H=256, one block (8 wave32) per row
// ---------------------------------------------------------------------------
__global__ __launch_bounds__(256)
void kan_layernorm(const float* __restrict__ in, float* __restrict__ out,
                   const float* __restrict__ gamma, const float* __restrict__ beta,
                   int H) {
    const int b = blockIdx.x, t = threadIdx.x;
    float v = in[(size_t)b * H + t];
    __shared__ float red[8];

    float s = v;
    for (int off = 16; off > 0; off >>= 1) s += __shfl_down(s, off);
    if ((t & 31) == 0) red[t >> 5] = s;
    __syncthreads();
    if (t < 8) {
        float r = red[t];
        for (int off = 4; off > 0; off >>= 1) r += __shfl_down(r, off);
        if (t == 0) red[0] = r;
    }
    __syncthreads();
    const float mu = red[0] / (float)H;
    __syncthreads();

    float d = v - mu;
    float q = d * d;
    for (int off = 16; off > 0; off >>= 1) q += __shfl_down(q, off);
    if ((t & 31) == 0) red[t >> 5] = q;
    __syncthreads();
    if (t < 8) {
        float r = red[t];
        for (int off = 4; off > 0; off >>= 1) r += __shfl_down(r, off);
        if (t == 0) red[0] = r;
    }
    __syncthreads();
    const float var = red[0] / (float)H;

    out[(size_t)b * H + t] = d * rsqrtf(var + LN_EPS) * gamma[t] + beta[t];
}

// ---------------------------------------------------------------------------
// 5. edge1 streamer: edge = silu(h)*bs + 2-tap hat spline; float4 NT stores
//    one block per batch row; 256 threads = 8 i-slices x 32 o-quads
// ---------------------------------------------------------------------------
__global__ __launch_bounds__(256)
void kan_edge1(const float* __restrict__ hln, const float* __restrict__ bs1,
               const float* __restrict__ w1, float* __restrict__ edge_out,
               double* __restrict__ acc, int H, int O) {
    const int b = blockIdx.x;
    const int t = threadIdx.x;
    const int o = (t & 31) * 4;       // 4 consecutive outputs per lane
    const int islice = t >> 5;        // 8 slices over i
    double ls = 0.0, lq = 0.0;

    for (int i = islice; i < H; i += 8) {
        float hv = hln[(size_t)b * H + i];
        float sv = silu(hv);
        float u  = (hv + GRANGE) / SPACING;
        float jf = floorf(u);
        int   j  = (int)jf;
        float tf = u - jf;
        const bool in0 = (j >= 0) && (j < NGRID);
        const bool in1 = (j + 1 >= 0) && (j + 1 < NGRID);
        v4f ev;
#pragma unroll
        for (int c = 0; c < 4; ++c) {
            const float* wrow = &w1[((size_t)i * O + o + c) * NGRID];
            float sp = 0.0f;
            if (in0) sp += wrow[j] * (1.0f - tf);
            if (in1) sp += wrow[j + 1] * tf;
            float e = sv * bs1[(size_t)i * O + o + c] + sp;
            ev[c] = e;
            ls += (double)e;
            lq += (double)e * (double)e;
        }
        __builtin_nontemporal_store(ev,
            (v4f*)&edge_out[((size_t)b * H + i) * O + o]);
    }

    __shared__ double rs[256];
    __shared__ double rq[256];
    rs[t] = ls; rq[t] = lq;
    __syncthreads();
    for (int off = 128; off > 0; off >>= 1) {
        if (t < off) { rs[t] += rs[t + off]; rq[t] += rq[t + off]; }
        __syncthreads();
    }
    if (t == 0) {
        atomicAdd(&acc[0], rs[0]);
        atomicAdd(&acc[1], rq[0]);
    }
}

// ---------------------------------------------------------------------------
// 6. finalize the 4 scalar diagnostics
// ---------------------------------------------------------------------------
__global__ void kan_finalize(const double* __restrict__ acc,
                             float* __restrict__ s) {
    const double nE = (double)BATCH * HID * OUTW;
    const double nB = (double)BATCH * HID * NGRID;
    double se_m = acc[0] / nE;
    double se_v = acc[1] / nE - se_m * se_m;
    double sb_m = acc[2] / nB;
    double sb_v = acc[3] / nB - sb_m * sb_m;
    s[0] = (float)se_m;
    s[1] = (float)sqrt(fmax(se_v, 0.0));
    s[2] = (float)sb_m;
    s[3] = (float)sqrt(fmax(sb_v, 0.0));
}

// ---------------------------------------------------------------------------
// launcher
// ---------------------------------------------------------------------------
extern "C" void kernel_launch(void* const* d_in, const int* in_sizes, int n_in,
                              void* d_out, int out_size, void* d_ws, size_t ws_size,
                              hipStream_t stream) {
    const float* x     = (const float*)d_in[0];
    const float* bsc0  = (const float*)d_in[1];
    const float* sw0   = (const float*)d_in[2];
    const float* bias0 = (const float*)d_in[3];
    const float* bsc1  = (const float*)d_in[4];
    const float* sw1   = (const float*)d_in[5];
    const float* bias1 = (const float*)d_in[6];
    const float* gam   = (const float*)d_in[7];
    const float* bet   = (const float*)d_in[8];

    // output layout: out | edge1 | basis1 | 4 scalars
    float* out     = (float*)d_out;
    float* edge1   = out   + (size_t)BATCH * OUTW;
    float* basis1  = edge1 + (size_t)BATCH * HID * OUTW;
    float* scalars = basis1 + (size_t)BATCH * HID * NGRID;

    // workspace layout
    char* ws = (char*)d_ws;
    double* acc  = (double*)ws;                                  size_t off = 256;
    __bf16* A0   = (__bf16*)(ws + off);  off += (size_t)BATCH * K0 * 2;
    __bf16* W0   = (__bf16*)(ws + off);  off += (size_t)K0 * HID * 2;
    float*  hpre = (float*) (ws + off);  off += (size_t)BATCH * HID * 4;
    float*  hln  = (float*) (ws + off);  off += (size_t)BATCH * HID * 4;
    __bf16* A1   = (__bf16*)(ws + off);  off += (size_t)BATCH * K1 * 2;
    __bf16* W1   = (__bf16*)(ws + off);  off += (size_t)K1 * OUTW * 2;
    (void)off; (void)ws_size;

    kan_init<<<1, 32, 0, stream>>>(acc);

    kan_build_W<<<(K0 * HID + 255) / 256, 256, 0, stream>>>(bsc0, sw0, W0, I0, HID, K0);
    kan_build_W<<<(K1 * OUTW + 255) / 256, 256, 0, stream>>>(bsc1, sw1, W1, HID, OUTW, K1);

    kan_build_A0<<<(BATCH * I0 + 255) / 256, 256, 0, stream>>>(x, A0, BATCH, I0, K0);

    const float scale0 = 1.0f / sqrtf((float)I0);   // 1/sqrt(128)
    kan_gemm_bf16<<<dim3(BATCH / 64, HID / 64), 256, 0, stream>>>(
        A0, W0, hpre, bias0, BATCH, HID, K0, scale0);

    kan_layernorm<<<BATCH, HID, 0, stream>>>(hpre, hln, gam, bet, HID);

    kan_build_A1<<<(BATCH * HID + 255) / 256, 256, 0, stream>>>(
        hln, A1, basis1, acc, BATCH, HID, K1);

    const float scale1 = 1.0f / sqrtf((float)HID);  // 1/16
    kan_gemm_bf16<<<dim3(BATCH / 64, OUTW / 64), 256, 0, stream>>>(
        A1, W1, out, bias1, BATCH, OUTW, K1, scale1);

    kan_edge1<<<BATCH, 256, 0, stream>>>(hln, bsc1, sw1, edge1, acc, HID, OUTW);

    kan_finalize<<<1, 1, 0, stream>>>(acc, scalars);
}
